// DecoderWithAttention_9620726743380
// MI455X (gfx1250) — matrix operand order repaired and checked by reading, not compiled, not run on hardware
//
#include <hip/hip_runtime.h>
#include <math.h>

// ---------------- problem constants ----------------
constexpr int B_ = 64, P_ = 196, ENC_ = 2048;
constexpr int A_ = 512, E_ = 512, D_ = 512;
constexpr int V_ = 10000, L_ = 52, T_ = 51;
constexpr int KC_ = E_ + ENC_ + D_;          // 3072 (concat LSTM input)

// output layout (floats, concatenated in reference return order)
constexpr long long OFF_CAPS = (long long)B_ * T_ * V_;             // B*L
constexpr long long OFF_DLEN = OFF_CAPS + (long long)B_ * L_;       // B
constexpr long long OFF_ALPH = OFF_DLEN + B_;                       // B*T*P
constexpr long long OFF_SIDX = OFF_ALPH + (long long)B_ * T_ * P_;  // B

// ---------------- WMMA types ----------------
typedef __attribute__((ext_vector_type(8)))  unsigned short v8u;
typedef __attribute__((ext_vector_type(16))) unsigned short v16u;
typedef __attribute__((ext_vector_type(16))) __bf16         v16bf;
typedef __attribute__((ext_vector_type(8)))  float          v8f;

__device__ __forceinline__ unsigned short f2bf(float f) {
    unsigned int u = __float_as_uint(f);
    u += 0x7FFFu + ((u >> 16) & 1u);   // round-to-nearest-even
    return (unsigned short)(u >> 16);
}
__device__ __forceinline__ float bf2f(unsigned short u) {
    return __uint_as_float(((unsigned int)u) << 16);
}

// A fragment (16x32 bf16): lane holds K = half*8+0..7 and 16+half*8+0..7 (16B-aligned b128 loads)
__device__ __forceinline__ v16bf load_a(const unsigned short* __restrict__ X, int row, int ld,
                                        int kb, int half) {
    const unsigned short* p = X + (size_t)row * ld + kb + half * 8;
    v8u lo = *(const v8u*)p;
    v8u hi = *(const v8u*)(p + 16);
    v16u a = __builtin_shufflevector(lo, hi, 0, 1, 2, 3, 4, 5, 6, 7,
                                     8, 9, 10, 11, 12, 13, 14, 15);
    return __builtin_bit_cast(v16bf, a);
}
// B fragment (32x16 bf16): lane holds K = half*8+0..7 and 16+half*8+0..7 (two b128 loads)
__device__ __forceinline__ v16bf load_b(const unsigned short* __restrict__ W, int n, int ld,
                                        int kb, int half) {
    const unsigned short* p = W + (size_t)n * ld + kb + half * 8;
    v8u lo = *(const v8u*)p;
    v8u hi = *(const v8u*)(p + 16);
    v16u b = __builtin_shufflevector(lo, hi, 0, 1, 2, 3, 4, 5, 6, 7,
                                     8, 9, 10, 11, 12, 13, 14, 15);
    return __builtin_bit_cast(v16bf, b);
}

__device__ __forceinline__ float sigmoidf_(float x) { return 1.f / (1.f + __expf(-x)); }

// ---------------- generic WMMA GEMM:  out = act(X @ W^T + bias) ----------------
// act 0: none; act 1: sigmoid(.)*mul[row*mulLd+col]; act 2: zero row unless t < dlens[row]
// Dual output: Outf (fp32) and/or Outb (bf16) — either may be null.
__global__ void wmma_gemm(const unsigned short* __restrict__ X,
                          const unsigned short* __restrict__ W, const float* __restrict__ bias,
                          float* __restrict__ Outf, unsigned short* __restrict__ Outb,
                          long long outStride, long long outBase,
                          int M, int N, int K, int act,
                          const float* __restrict__ mul, int mulLd,
                          const int* __restrict__ dlens, int t) {
    int wid  = (blockIdx.x * blockDim.x + threadIdx.x) >> 5;
    int lane = threadIdx.x & 31;
    int tilesN = N >> 4;
    if (wid >= (M >> 4) * tilesN) return;           // wave-uniform: EXEC stays all-ones
    int mt = (wid / tilesN) << 4;
    int nt = (wid % tilesN) << 4;
    int half = lane >> 4, l15 = lane & 15;

    v8f acc = {0.f, 0.f, 0.f, 0.f, 0.f, 0.f, 0.f, 0.f};
    for (int kb = 0; kb < K; kb += 32) {
        v16bf a = load_a(X, mt + l15, K, kb, half);
        v16bf b = load_b(W, nt + l15, K, kb, half);
        acc = __builtin_amdgcn_wmma_f32_16x16x32_bf16(false, a, false, b,
                                                      (short)0, acc, false, false);
    }
    int col = nt + l15;
    float bs = bias ? bias[col] : 0.f;
#pragma unroll
    for (int r = 0; r < 8; ++r) {
        int row = mt + r + (half << 3);
        float v = acc[r] + bs;
        if (act == 1) v = sigmoidf_(v) * mul[(size_t)row * mulLd + col];
        if (act == 2 && !(t < dlens[row])) v = 0.f;
        long long idx = outBase + (long long)row * outStride + col;
        if (Outf) Outf[idx] = v;
        if (Outb) Outb[idx] = f2bf(v);
    }
}

// ---------------- setup kernels ----------------
__global__ void k_sort(const int* __restrict__ cl, const int* __restrict__ caps,
                       int* __restrict__ sidx, int* __restrict__ dlens,
                       int* __restrict__ caps_s, float* __restrict__ out) {
    __shared__ int s_cl[B_], s_si[B_];
    int b = threadIdx.x;
    if (b < B_) s_cl[b] = cl[b];
    __syncthreads();
    if (b < B_) {
        int my = s_cl[b], rank = 0;
        for (int j = 0; j < B_; ++j) {
            int o = s_cl[j];
            if (o > my || (o == my && j < b)) rank++;    // stable argsort(-cl)
        }
        s_si[rank] = b;
    }
    __syncthreads();
    if (b < B_) {
        int s = s_si[b];
        sidx[b] = s;
        int dl = s_cl[s] - 1;
        dlens[b] = dl;
        out[OFF_DLEN + b] = (float)dl;
        out[OFF_SIDX + b] = (float)s;
        for (int t = 0; t < L_; ++t) {
            int tok = caps[s * L_ + t];
            caps_s[b * L_ + t] = tok;
            out[OFF_CAPS + b * L_ + t] = (float)tok;
        }
    }
}

__global__ void k_cast_bf16(const float* __restrict__ s, unsigned short* __restrict__ d, int n) {
    int i = blockIdx.x * blockDim.x + threadIdx.x;
    if (i < n) d[i] = f2bf(s[i]);
}

// Wc[n][0:2560]=w_ih[n], Wc[n][2560:3072]=w_hh[n]
__global__ void k_build_wc(const float* __restrict__ wih, const float* __restrict__ whh,
                           unsigned short* __restrict__ wc) {
    int i = blockIdx.x * blockDim.x + threadIdx.x;
    if (i >= 4 * D_ * KC_) return;
    int n = i / KC_, k = i % KC_;
    float v = (k < E_ + ENC_) ? wih[(size_t)n * (E_ + ENC_) + k]
                              : whh[(size_t)n * D_ + (k - (E_ + ENC_))];
    wc[i] = f2bf(v);
}

__global__ void k_bias_c(const float* __restrict__ bih, const float* __restrict__ bhh,
                         float* __restrict__ bc) {
    int i = blockIdx.x * blockDim.x + threadIdx.x;
    if (i < 4 * D_) bc[i] = bih[i] + bhh[i];
}

__global__ void k_gather_enc(const float* __restrict__ enc, const int* __restrict__ sidx,
                             unsigned short* __restrict__ enc_s) {
    long long i = (long long)blockIdx.x * blockDim.x + threadIdx.x;
    long long n = (long long)B_ * P_ * ENC_;
    if (i >= n) return;
    int b = (int)(i / ((long long)P_ * ENC_));
    long long rem = i % ((long long)P_ * ENC_);
    enc_s[i] = f2bf(enc[(long long)sidx[b] * P_ * ENC_ + rem]);
}

__global__ void k_mean_enc(const float* __restrict__ enc, const int* __restrict__ sidx,
                           unsigned short* __restrict__ mean_bf) {
    int i = blockIdx.x * blockDim.x + threadIdx.x;
    if (i >= B_ * ENC_) return;
    int b = i / ENC_, e = i % ENC_;
    const float* src = enc + (long long)sidx[b] * P_ * ENC_ + e;
    float s = 0.f;
    for (int p = 0; p < P_; ++p) s += src[(size_t)p * ENC_];
    mean_bf[i] = f2bf(s * (1.f / P_));
}

__global__ void k_gather_emb(const float* __restrict__ emb, const int* __restrict__ caps_s,
                             unsigned short* __restrict__ emb_sb) {
    int i = blockIdx.x * blockDim.x + threadIdx.x;
    if (i >= B_ * T_ * E_) return;
    int e = i % E_, bt = i / E_;
    int b = bt / T_, t = bt % T_;
    int tok = caps_s[b * L_ + t];
    emb_sb[i] = f2bf(emb[(size_t)tok * E_ + e]);
}

// ---------------- per-step kernels ----------------
__global__ void k_scores(const float* __restrict__ att1, const float* __restrict__ att2,
                         const float* __restrict__ faw, const float* __restrict__ fab,
                         float* __restrict__ scores) {
    int wid  = (blockIdx.x * blockDim.x + threadIdx.x) >> 5;
    int lane = threadIdx.x & 31;
    if (wid >= B_ * P_) return;
    int b = wid / P_;
    const float* a1 = att1 + (size_t)wid * A_;
    const float* a2 = att2 + (size_t)b * A_;
    float s = 0.f;
    for (int i = lane; i < A_; i += 32) {
        float v = a1[i] + a2[i];
        v = v > 0.f ? v : 0.f;
        s += v * faw[i];
    }
#pragma unroll
    for (int off = 16; off > 0; off >>= 1) s += __shfl_xor(s, off, 32);
    if (lane == 0) scores[wid] = s + fab[0];
}

__global__ void k_softmax(const float* __restrict__ scores, float* __restrict__ alpha,
                          float* __restrict__ out, const int* __restrict__ dlens, int t) {
    __shared__ float sh[256];
    int b = blockIdx.x, tid = threadIdx.x;
    float v = (tid < P_) ? scores[b * P_ + tid] : -1e30f;
    sh[tid] = v; __syncthreads();
    for (int s = 128; s > 0; s >>= 1) {
        if (tid < s) sh[tid] = fmaxf(sh[tid], sh[tid + s]);
        __syncthreads();
    }
    float mx = sh[0]; __syncthreads();
    float e = (tid < P_) ? __expf(v - mx) : 0.f;
    sh[tid] = e; __syncthreads();
    for (int s = 128; s > 0; s >>= 1) {
        if (tid < s) sh[tid] += sh[tid + s];
        __syncthreads();
    }
    float inv = 1.f / sh[0];
    if (tid < P_) {
        float a = e * inv;
        alpha[b * P_ + tid] = a;
        bool m = t < dlens[b];
        out[OFF_ALPH + (long long)b * T_ * P_ + (long long)t * P_ + tid] = m ? a : 0.f;
    }
}

__global__ void k_awe(const unsigned short* __restrict__ enc_s, const float* __restrict__ alpha,
                      float* __restrict__ awe) {
    int i = blockIdx.x * blockDim.x + threadIdx.x;
    if (i >= B_ * ENC_) return;
    int b = i / ENC_, e = i % ENC_;
    const unsigned short* src = enc_s + (long long)b * P_ * ENC_ + e;
    const float* al = alpha + b * P_;
    float s = 0.f;
    for (int p = 0; p < P_; ++p) s += al[p] * bf2f(src[(size_t)p * ENC_]);
    awe[i] = s;
}

// xin_bf[b][0:512]=e_t ; xin_bf[b][2560:3072]=h_bf  (cols 512:2560 from f_beta GEMM epilogue)
__global__ void k_prep_xin(const unsigned short* __restrict__ emb_sb,
                           const unsigned short* __restrict__ h_bf,
                           unsigned short* __restrict__ xin_bf, int t) {
    int i = blockIdx.x * blockDim.x + threadIdx.x;
    if (i >= B_ * (E_ + D_)) return;
    int b = i / (E_ + D_), j = i % (E_ + D_);
    if (j < E_) xin_bf[(size_t)b * KC_ + j] = emb_sb[((size_t)b * T_ + t) * E_ + j];
    else        xin_bf[(size_t)b * KC_ + 2560 + (j - E_)] = h_bf[(size_t)b * D_ + (j - E_)];
}

__global__ void k_lstm(const float* __restrict__ gates, float* __restrict__ h,
                       unsigned short* __restrict__ h_bf, float* __restrict__ c,
                       const int* __restrict__ dlens, int t) {
    int i = blockIdx.x * blockDim.x + threadIdx.x;
    if (i >= B_ * D_) return;
    int b = i / D_;
    int j = i % D_;
    const float* g = gates + (size_t)b * 4 * D_;
    float gi = sigmoidf_(g[j]);
    float gf = sigmoidf_(g[D_ + j]);
    float gg = tanhf(g[2 * D_ + j]);
    float go = sigmoidf_(g[3 * D_ + j]);
    float cn = gf * c[i] + gi * gg;
    float hn = go * tanhf(cn);
    if (t < dlens[b]) { h[i] = hn; c[i] = cn; }
    h_bf[i] = f2bf(h[i]);              // keep bf16 mirror in sync every step
}

// ---------------- host ----------------
static inline int cdiv(long long a, int b) { return (int)((a + b - 1) / b); }

extern "C" void kernel_launch(void* const* d_in, const int* in_sizes, int n_in,
                              void* d_out, int out_size, void* d_ws, size_t ws_size,
                              hipStream_t stream) {
    const float* enc       = (const float*)d_in[0];
    const float* embedding = (const float*)d_in[1];
    const float* enc_att_w = (const float*)d_in[2];
    const float* enc_att_b = (const float*)d_in[3];
    const float* dec_att_w = (const float*)d_in[4];
    const float* dec_att_b = (const float*)d_in[5];
    const float* full_att_w= (const float*)d_in[6];
    const float* full_att_b= (const float*)d_in[7];
    const float* init_h_w  = (const float*)d_in[8];
    const float* init_h_b  = (const float*)d_in[9];
    const float* init_c_w  = (const float*)d_in[10];
    const float* init_c_b  = (const float*)d_in[11];
    const float* f_beta_w  = (const float*)d_in[12];
    const float* f_beta_b  = (const float*)d_in[13];
    const float* w_ih      = (const float*)d_in[14];
    const float* w_hh      = (const float*)d_in[15];
    const float* b_ih      = (const float*)d_in[16];
    const float* b_hh      = (const float*)d_in[17];
    const float* fc_w      = (const float*)d_in[18];
    const float* fc_b      = (const float*)d_in[19];
    const int*   caps      = (const int*)d_in[20];
    const int*   cap_lens  = (const int*)d_in[21];
    float* out = (float*)d_out;

    // workspace carve-out (256B aligned slots)
    char* w = (char*)d_ws;
    size_t o = 0;
    auto alloc = [&](size_t bytes) { void* p = w + o; o = (o + bytes + 255) & ~(size_t)255; return p; };
    int*            ws_sidx  = (int*)alloc(B_ * 4);
    int*            ws_dlens = (int*)alloc(B_ * 4);
    int*            ws_capss = (int*)alloc(B_ * L_ * 4);
    unsigned short* ws_encs  = (unsigned short*)alloc((size_t)B_ * P_ * ENC_ * 2);
    float*          ws_att1  = (float*)alloc((size_t)B_ * P_ * A_ * 4);
    unsigned short* ws_wenc  = (unsigned short*)alloc((size_t)A_ * ENC_ * 2);
    unsigned short* ws_wdec  = (unsigned short*)alloc((size_t)A_ * D_ * 2);
    unsigned short* ws_wfb   = (unsigned short*)alloc((size_t)ENC_ * D_ * 2);
    unsigned short* ws_wfc   = (unsigned short*)alloc((size_t)V_ * D_ * 2);
    unsigned short* ws_wih0  = (unsigned short*)alloc((size_t)D_ * ENC_ * 2);
    unsigned short* ws_wic0  = (unsigned short*)alloc((size_t)D_ * ENC_ * 2);
    unsigned short* ws_wc    = (unsigned short*)alloc((size_t)4 * D_ * KC_ * 2);
    float*          ws_bc    = (float*)alloc(4 * D_ * 4);
    unsigned short* ws_embsb = (unsigned short*)alloc((size_t)B_ * T_ * E_ * 2);
    unsigned short* ws_meanb = (unsigned short*)alloc((size_t)B_ * ENC_ * 2);
    float*          ws_h     = (float*)alloc((size_t)B_ * D_ * 4);
    unsigned short* ws_hbf   = (unsigned short*)alloc((size_t)B_ * D_ * 2);
    float*          ws_c     = (float*)alloc((size_t)B_ * D_ * 4);
    float*          ws_att2  = (float*)alloc((size_t)B_ * A_ * 4);
    float*          ws_scr   = (float*)alloc((size_t)B_ * P_ * 4);
    float*          ws_alpha = (float*)alloc((size_t)B_ * P_ * 4);
    float*          ws_awe   = (float*)alloc((size_t)B_ * ENC_ * 4);
    unsigned short* ws_xinb  = (unsigned short*)alloc((size_t)B_ * KC_ * 2);
    float*          ws_gates = (float*)alloc((size_t)B_ * 4 * D_ * 4);
    (void)ws_size; (void)n_in; (void)in_sizes; (void)out_size;

    auto gemm = [&](const unsigned short* X, const unsigned short* W, const float* bias,
                    float* Outf, unsigned short* Outb, long long stride, long long base,
                    int M, int N, int K, int act, const float* mul, int mulLd, int t) {
        long long waves = (long long)(M / 16) * (N / 16);
        int blocks = cdiv(waves * 32, 256);
        wmma_gemm<<<blocks, 256, 0, stream>>>(X, W, bias, Outf, Outb, stride, base,
                                              M, N, K, act, mul, mulLd, ws_dlens, t);
    };

    // ---- one-time setup ----
    k_sort<<<1, 64, 0, stream>>>(cap_lens, caps, ws_sidx, ws_dlens, ws_capss, out);
    k_cast_bf16<<<cdiv((size_t)A_ * ENC_, 256), 256, 0, stream>>>(enc_att_w, ws_wenc, A_ * ENC_);
    k_cast_bf16<<<cdiv((size_t)A_ * D_, 256), 256, 0, stream>>>(dec_att_w, ws_wdec, A_ * D_);
    k_cast_bf16<<<cdiv((size_t)ENC_ * D_, 256), 256, 0, stream>>>(f_beta_w, ws_wfb, ENC_ * D_);
    k_cast_bf16<<<cdiv((size_t)V_ * D_, 256), 256, 0, stream>>>(fc_w, ws_wfc, V_ * D_);
    k_cast_bf16<<<cdiv((size_t)D_ * ENC_, 256), 256, 0, stream>>>(init_h_w, ws_wih0, D_ * ENC_);
    k_cast_bf16<<<cdiv((size_t)D_ * ENC_, 256), 256, 0, stream>>>(init_c_w, ws_wic0, D_ * ENC_);
    k_build_wc<<<cdiv((size_t)4 * D_ * KC_, 256), 256, 0, stream>>>(w_ih, w_hh, ws_wc);
    k_bias_c<<<cdiv(4 * D_, 256), 256, 0, stream>>>(b_ih, b_hh, ws_bc);
    k_gather_enc<<<cdiv((long long)B_ * P_ * ENC_, 256), 256, 0, stream>>>(enc, ws_sidx, ws_encs);
    k_mean_enc<<<cdiv(B_ * ENC_, 256), 256, 0, stream>>>(enc, ws_sidx, ws_meanb);
    k_gather_emb<<<cdiv(B_ * T_ * E_, 256), 256, 0, stream>>>(embedding, ws_capss, ws_embsb);

    // h0 (fp32 + bf16 mirror) / c0 : mean_enc @ init_{h,c}_w^T -> 64x512
    gemm(ws_meanb, ws_wih0, init_h_b, ws_h, ws_hbf, D_, 0, B_, D_, ENC_, 0, nullptr, 0, 0);
    gemm(ws_meanb, ws_wic0, init_c_b, ws_c, nullptr, D_, 0, B_, D_, ENC_, 0, nullptr, 0, 0);
    // att1 : enc_s(12544x2048 bf16) @ enc_att_w^T -> 12544x512 fp32
    gemm(ws_encs, ws_wenc, enc_att_b, ws_att1, nullptr, A_, 0, B_ * P_, A_, ENC_, 0, nullptr, 0, 0);

    // ---- sequential decode ----
    for (int t = 0; t < T_; ++t) {
        // att2 = h @ dec_att_w^T + b
        gemm(ws_hbf, ws_wdec, dec_att_b, ws_att2, nullptr, A_, 0, B_, A_, D_, 0, nullptr, 0, t);
        // scores = relu(att1+att2) . full_att_w + b
        k_scores<<<cdiv((long long)B_ * P_ * 32, 256), 256, 0, stream>>>(
            ws_att1, ws_att2, full_att_w, full_att_b, ws_scr);
        // softmax over P, write alphas output (masked)
        k_softmax<<<B_, 256, 0, stream>>>(ws_scr, ws_alpha, out, ws_dlens, t);
        // awe = sum_p alpha * enc
        k_awe<<<cdiv(B_ * ENC_, 256), 256, 0, stream>>>(ws_encs, ws_alpha, ws_awe);
        // gated awe -> bf16 xin[:,512:2560] : sigmoid(h @ f_beta^T + b) * awe
        gemm(ws_hbf, ws_wfb, f_beta_b, nullptr, ws_xinb, KC_, 512, B_, ENC_, D_, 1, ws_awe, ENC_, t);
        // xin[:,0:512]=e_t, xin[:,2560:3072]=h (both bf16)
        k_prep_xin<<<cdiv(B_ * (E_ + D_), 256), 256, 0, stream>>>(ws_embsb, ws_hbf, ws_xinb, t);
        // gates = xin(64x3072) @ [w_ih|w_hh]^T + (b_ih+b_hh)
        gemm(ws_xinb, ws_wc, ws_bc, ws_gates, nullptr, 4 * D_, 0, B_, 4 * D_, KC_, 0, nullptr, 0, t);
        // LSTM pointwise, masked h/c update + bf16 h mirror
        k_lstm<<<cdiv(B_ * D_, 256), 256, 0, stream>>>(ws_gates, ws_h, ws_hbf, ws_c, ws_dlens, t);
        // preds = mask ? h @ fc_w^T + fc_b : 0  -> out[b][t][:]
        gemm(ws_hbf, ws_wfc, fc_b, out, nullptr, (long long)T_ * V_, (long long)t * V_,
             B_, V_, D_, 2, nullptr, 0, t);
    }
}
